// SymHQLinear_40252433498283
// MI455X (gfx1250) — matrix operand order repaired and checked by reading, not compiled, not run on hardware
//
#include <hip/hip_runtime.h>

// ---------------------------------------------------------------------------
// SymHQLinear: out[M,4096] = x[M,4096] @ W^T, W dequantized from codebook
// M = 4*2048 = 8192, OUT = IN = 4096.
// Compute-bound (AI ~1000 FLOP/B) -> bf16 WMMA matrix pipe with 3-term
// hi/lo split for ~f32 accuracy. Double-buffered LDS dequant staging.
// ---------------------------------------------------------------------------

typedef __bf16 bf16;
typedef __attribute__((ext_vector_type(4)))  __bf16 bf16x4;
typedef __attribute__((ext_vector_type(8)))  __bf16 bf16x8;
typedef __attribute__((ext_vector_type(16))) __bf16 bf16x16;
typedef __attribute__((ext_vector_type(8)))  float  v8f;
typedef __attribute__((ext_vector_type(4)))  float  f32x4;

#define IN_DIM   4096
#define OUT_DIM  4096
#define BM       128
#define BN       128
#define BK       32
#define NKT      (IN_DIM / BK)     // 128 K-steps
#define LDSS     40                // LDS row stride in bf16 (BK + 8 pad, keeps 16B align)

static __device__ __forceinline__ bf16x16 join8(bf16x8 a, bf16x8 b) {
    return __builtin_shufflevector(a, b, 0, 1, 2, 3, 4, 5, 6, 7,
                                         8, 9, 10, 11, 12, 13, 14, 15);
}

__global__ __launch_bounds__(256, 1)
void symhq_wmma_bf16x3_kernel(const float* __restrict__ x,
                              const float* __restrict__ codebook,
                              const float* __restrict__ scales,
                              const int*   __restrict__ indices,
                              const int*   __restrict__ signs,
                              float*       __restrict__ out)
{
    // hi/lo bf16 planes for X (activations) and W (dequantized weights),
    // double buffered: 8 * 128*40*2B = 80 KB of the 320 KB WGP LDS.
    __shared__ __align__(16) bf16 sXH[2][BM * LDSS];
    __shared__ __align__(16) bf16 sXL[2][BM * LDSS];
    __shared__ __align__(16) bf16 sWH[2][BN * LDSS];
    __shared__ __align__(16) bf16 sWL[2][BN * LDSS];

    const int tid  = threadIdx.x;
    const int lane = tid & 31;
    const int wave = tid >> 5;
    const int wm   = wave & 1;    // 2 waves along M: 64 rows each
    const int wn   = wave >> 1;   // 4 waves along N: 32 cols each
    const int lr   = lane & 15;   // position within 16 (row for A, col for B/C)
    const int lh   = lane >> 4;   // lane half

    const int m0 = blockIdx.y * BM;
    const int n0 = blockIdx.x * BN;

    // ---- register staging for the next K-tile (global -> regs -> LDS) ----
    f32x4 xs[4];          // 16 floats of X per thread
    f32x4 cw[2][2];       // 2 codewords (8 f32 each) per thread
    float wsc[2];
    int   wsg[2];

    auto fetch = [&](int kt) {
        const int k0 = kt * BK;
        // X tile: 128 rows x 32 cols f32 = 1024 float4s; 4 per thread.
        #pragma unroll
        for (int i = 0; i < 4; ++i) {
            int lin = tid + 256 * i;
            int row = lin >> 3;        // 0..127
            int c4  = lin & 7;         // 0..7 float4s in the row
            xs[i] = *(const f32x4*)(x + (size_t)(m0 + row) * IN_DIM + k0 + c4 * 4);
        }
        // W tile: 128 rows x 4 codeword-groups (8 k each); 2 per thread.
        const int kg0 = k0 >> 3;
        #pragma unroll
        for (int i = 0; i < 2; ++i) {
            int lin = tid + 256 * i;
            int n   = lin >> 2;        // 0..127
            int g   = lin & 3;         // group within the K-tile
            int gi  = (n0 + n) * (IN_DIM / 8) + kg0 + g;
            int idx = indices[gi];
            wsg[i]  = signs[gi];
            wsc[i]  = scales[n0 + n];
            const f32x4* cb = (const f32x4*)codebook + (size_t)idx * 2;
            cw[i][0] = cb[0];
            cw[i][1] = cb[1];
        }
    };

    auto stage = [&](int buf) {
        bf16* XH = sXH[buf]; bf16* XL = sXL[buf];
        bf16* WH = sWH[buf]; bf16* WL = sWL[buf];
        #pragma unroll
        for (int i = 0; i < 4; ++i) {
            int lin = tid + 256 * i;
            int row = lin >> 3;
            int c4  = lin & 7;
            bf16x4 h, l;
            #pragma unroll
            for (int j = 0; j < 4; ++j) {
                float v  = xs[i][j];
                bf16  hb = (bf16)v;
                h[j] = hb;
                l[j] = (bf16)(v - (float)hb);
            }
            *(bf16x4*)(XH + row * LDSS + c4 * 4) = h;
            *(bf16x4*)(XL + row * LDSS + c4 * 4) = l;
        }
        #pragma unroll
        for (int i = 0; i < 2; ++i) {
            int lin = tid + 256 * i;
            int n   = lin >> 2;
            int g   = lin & 3;
            int s   = wsg[i];
            float sc = wsc[i];
            bf16x8 h, l;
            #pragma unroll
            for (int j = 0; j < 8; ++j) {
                float v = cw[i][j >> 2][j & 3] * sc;
                v = ((s >> j) & 1) ? v : -v;     // sign bit: 1 -> +, 0 -> -
                bf16 hb = (bf16)v;
                h[j] = hb;
                l[j] = (bf16)(v - (float)hb);
            }
            *(bf16x8*)(WH + n * LDSS + g * 8) = h;
            *(bf16x8*)(WL + n * LDSS + g * 8) = l;
        }
    };

    // ---- accumulators: 4 (M subtiles) x 2 (N subtiles) of 16x16 f32 ----
    v8f acc[4][2];
    #pragma unroll
    for (int ms = 0; ms < 4; ++ms)
        #pragma unroll
        for (int ns = 0; ns < 2; ++ns)
            acc[ms][ns] = v8f{0.f, 0.f, 0.f, 0.f, 0.f, 0.f, 0.f, 0.f};

    auto compute = [&](int buf) {
        const bf16* XH = sXH[buf]; const bf16* XL = sXL[buf];
        const bf16* WH = sWH[buf]; const bf16* WL = sWL[buf];
        // B fragments: chunks at k = 8*lh and k = 16 + 8*lh, col = lr.
        bf16x16 bh[2], bl[2];
        const int koff = lh * 8;
        #pragma unroll
        for (int ns = 0; ns < 2; ++ns) {
            int n = wn * 32 + ns * 16 + lr;
            bh[ns] = join8(*(const bf16x8*)(WH + n * LDSS + koff),
                           *(const bf16x8*)(WH + n * LDSS + koff + 16));
            bl[ns] = join8(*(const bf16x8*)(WL + n * LDSS + koff),
                           *(const bf16x8*)(WL + n * LDSS + koff + 16));
        }
        #pragma unroll
        for (int ms = 0; ms < 4; ++ms) {
            // A fragment: chunks at k = 8*lh and k = 16 + 8*lh, row = lr.
            int m = wm * 64 + ms * 16 + lr;
            bf16x16 ah = join8(*(const bf16x8*)(XH + m * LDSS + 8 * lh),
                               *(const bf16x8*)(XH + m * LDSS + 16 + 8 * lh));
            bf16x16 al = join8(*(const bf16x8*)(XL + m * LDSS + 8 * lh),
                               *(const bf16x8*)(XL + m * LDSS + 16 + 8 * lh));
            #pragma unroll
            for (int ns = 0; ns < 2; ++ns) {
                // 3-term bf16 split: hi*hi + hi*lo + lo*hi (f32 accumulate)
                acc[ms][ns] = __builtin_amdgcn_wmma_f32_16x16x32_bf16(
                    false, ah, false, bh[ns], (short)0, acc[ms][ns], false, false);
                acc[ms][ns] = __builtin_amdgcn_wmma_f32_16x16x32_bf16(
                    false, ah, false, bl[ns], (short)0, acc[ms][ns], false, false);
                acc[ms][ns] = __builtin_amdgcn_wmma_f32_16x16x32_bf16(
                    false, al, false, bh[ns], (short)0, acc[ms][ns], false, false);
            }
        }
    };

    // ---- main pipeline: double-buffered LDS, one barrier per K-step ----
    fetch(0);
    stage(0);
    __syncthreads();

    for (int kt = 0; kt < NKT; ++kt) {
        const int cur = kt & 1;
        const bool more = (kt + 1) < NKT;
        if (more) fetch(kt + 1);     // global loads overlap the WMMAs below
        compute(cur);
        if (more) stage(cur ^ 1);    // write other buffer; prior readers synced
        __syncthreads();
    }

    // ---- epilogue: C/D layout -> row = vgpr + 8*(lane/16), col = lr ----
    #pragma unroll
    for (int ms = 0; ms < 4; ++ms) {
        #pragma unroll
        for (int ns = 0; ns < 2; ++ns) {
            int n = n0 + wn * 32 + ns * 16 + lr;
            #pragma unroll
            for (int r = 0; r < 8; ++r) {
                int m = m0 + wm * 64 + ms * 16 + r + 8 * lh;
                out[(size_t)m * OUT_DIM + n] = acc[ms][ns][r];
            }
        }
    }
}

extern "C" void kernel_launch(void* const* d_in, const int* in_sizes, int n_in,
                              void* d_out, int out_size, void* d_ws, size_t ws_size,
                              hipStream_t stream) {
    const float* x        = (const float*)d_in[0];
    const float* codebook = (const float*)d_in[1];
    const float* scales   = (const float*)d_in[2];
    const int*   indices  = (const int*)d_in[3];
    const int*   signs    = (const int*)d_in[4];
    float*       out      = (float*)d_out;

    const int Mtot = in_sizes[0] / IN_DIM;   // 8192
    dim3 grid(OUT_DIM / BN, Mtot / BM);      // 32 x 64 blocks
    symhq_wmma_bf16x3_kernel<<<grid, dim3(256), 0, stream>>>(
        x, codebook, scales, indices, signs, out);
}
